// GCN_34746285424663
// MI455X (gfx1250) — matrix operand order repaired and checked by reading, not compiled, not run on hardware
//
#include <hip/hip_runtime.h>
#include <hip/hip_bf16.h>
#include <math.h>

// ---------------------------------------------------------------------------
// CDNA5 WMMA types
// ---------------------------------------------------------------------------
typedef __attribute__((ext_vector_type(16))) __bf16 v16bf;
typedef __attribute__((ext_vector_type(8)))  float  v8f;

union FragBF { v16bf v; unsigned int u[8]; };

static __device__ __forceinline__ unsigned int pack_bf2(float lo, float hi) {
    unsigned short a = __builtin_bit_cast(unsigned short, (__bf16)lo);
    unsigned short b = __builtin_bit_cast(unsigned short, (__bf16)hi);
    return (unsigned int)a | ((unsigned int)b << 16);
}

// ---------------------------------------------------------------------------
// GCN normalization: deg -> dinv = rsqrt(deg)
// ---------------------------------------------------------------------------
__global__ void gcn_deg_init(float* deg, int N) {
    int i = blockIdx.x * blockDim.x + threadIdx.x;
    if (i < N) deg[i] = 1.0f;  // self-loop contribution
}

__global__ void gcn_deg_count(const int* __restrict__ dst, float* deg, int E) {
    int e = blockIdx.x * blockDim.x + threadIdx.x;
    if (e < E) unsafeAtomicAdd(&deg[dst[e]], 1.0f);
}

__global__ void gcn_rsqrt_inplace(float* d, int N) {
    int i = blockIdx.x * blockDim.x + threadIdx.x;
    if (i < N) {
        float v = d[i];
        d[i] = v > 0.0f ? rsqrtf(v) : 0.0f;
    }
}

// ---------------------------------------------------------------------------
// Weight convert f32 -> bf16, transposed: WT[n*K + k] = W[k*KOUT + n]
// ---------------------------------------------------------------------------
__global__ void gcn_w_convT(const float* __restrict__ W, __bf16* __restrict__ WT,
                            int K, int KOUT) {
    int idx = blockIdx.x * blockDim.x + threadIdx.x;
    int total = K * KOUT;
    if (idx < total) {
        int k = idx / KOUT;
        int n = idx - k * KOUT;
        WT[(size_t)n * K + k] = (__bf16)W[idx];
    }
}

// ---------------------------------------------------------------------------
// Tiled bf16 WMMA GEMM: C[N x KOUT] = A[N x K] * W[K x KOUT] (+ bias)
// Block: 256 threads = 8 waves; block tile 128 rows x KOUT cols.
// Wave w owns rows [w*16, w*16+16) x all KOUT/16 column tiles.
// ---------------------------------------------------------------------------
template <int KOUT>
__global__ __launch_bounds__(256) void gcn_gemm_bf16(
    const float* __restrict__ A, const __bf16* __restrict__ WT,
    const float* __restrict__ bias, float* __restrict__ C,
    int N, int K)
{
    constexpr int NT = KOUT / 16;
    __shared__ unsigned int ldsA[128][17];   // 128 rows x 32 bf16 (16 uints) +pad
    __shared__ unsigned int ldsB[KOUT][17];  // KOUT cols x 32 bf16 (16 uints) +pad

    const int t    = threadIdx.x;
    const int wave = t >> 5;
    const int lane = t & 31;
    const int h    = lane >> 4;   // lane half (ISA 16-bit A/B layout)
    const int l16  = lane & 15;
    const int rowBase = blockIdx.x * 128;

    v8f acc[NT];
    for (int i = 0; i < NT; ++i) acc[i] = (v8f){0,0,0,0,0,0,0,0};

    const int rload = t >> 1;   // 0..127: A row staged by this thread
    const int halfc = t & 1;    // 16-col half of the 32-wide K chunk

    for (int kk = 0; kk < K; kk += 32) {
        // ---- stage A tile, converting f32 -> packed bf16 pairs -------------
        {
            int gr = rowBase + rload;
            if (gr >= N) gr = N - 1;  // clamp: stores are guarded later
            const float* ap = A + (size_t)gr * K + kk + halfc * 16;
            #pragma unroll
            for (int j = 0; j < 4; ++j) {
                float4 v = *(const float4*)(ap + j * 4);
                ldsA[rload][halfc * 8 + j * 2 + 0] = pack_bf2(v.x, v.y);
                ldsA[rload][halfc * 8 + j * 2 + 1] = pack_bf2(v.z, v.w);
            }
        }
        // ---- stage B tile from pre-transposed bf16 weights -----------------
        for (int p = t; p < 16 * KOUT; p += 256) {
            int k2 = p & 15;     // packed k-pair index 0..15
            int n  = p >> 4;     // output column 0..KOUT-1
            ldsB[n][k2] = *(const unsigned int*)(WT + (size_t)n * K + kk + 2 * k2);
        }
        __syncthreads();

        // ---- build A fragment (ISA 16-bit A 16x32 layout) ------------------
        FragBF a;
        #pragma unroll
        for (int r = 0; r < 8; ++r) {
            // k0 = (r<4?0:16) + h*8 + (r&3)*2  -> packed uint idx = k0/2
            int idx = ((r >> 2) << 3) + h * 4 + (r & 3);
            a.u[r] = ldsA[wave * 16 + l16][idx];
        }

        // ---- per column tile: build B fragment, issue WMMA -----------------
        #pragma unroll
        for (int ct = 0; ct < NT; ++ct) {
            FragBF b;
            #pragma unroll
            for (int r = 0; r < 8; ++r)           // k = (r<4?0:16) + h*8 + (r&3)*2
                b.u[r] = ldsB[ct * 16 + l16][((r >> 2) << 3) + h * 4 + (r & 3)];
            acc[ct] = __builtin_amdgcn_wmma_f32_16x16x32_bf16(
                false, a.v, false, b.v, (short)0, acc[ct], false, false);
        }
        __syncthreads();
    }

    // ---- store C per documented f32 C/D layout: M = r + 8*h, N = l16 -------
    #pragma unroll
    for (int ct = 0; ct < NT; ++ct) {
        int col = ct * 16 + l16;
        float badd = bias ? bias[col] : 0.0f;
        #pragma unroll
        for (int r = 0; r < 8; ++r) {
            int g = rowBase + wave * 16 + r + 8 * h;
            if (g < N) C[(size_t)g * KOUT + col] = acc[ct][r] + badd;
        }
    }
}

// ---------------------------------------------------------------------------
// Aggregation: out = bias + dinv^2 * h (self loop), then edge scatter-add
// Hdim fixed = 128 (4 floats per lane, one wave per edge)
// ---------------------------------------------------------------------------
__global__ void gcn_agg_init(const float* __restrict__ h,
                             const float* __restrict__ dinv,
                             const float* __restrict__ bias,
                             float* __restrict__ out, int N) {
    int idx = blockIdx.x * blockDim.x + threadIdx.x;
    int total = N * 128;
    if (idx < total) {
        int i = idx >> 7;
        int f = idx & 127;
        float di = dinv[i];
        out[idx] = bias[f] + di * di * h[idx];
    }
}

__global__ __launch_bounds__(256) void gcn_agg_edges(
    const float* __restrict__ h, const int* __restrict__ src,
    const int* __restrict__ dst, const float* __restrict__ dinv,
    float* __restrict__ out, int E)
{
    int gwave = (blockIdx.x * blockDim.x + threadIdx.x) >> 5;
    int lane  = threadIdx.x & 31;
    int nwave = (gridDim.x * blockDim.x) >> 5;
    for (int e = gwave; e < E; e += nwave) {
        int s = src[e], d = dst[e];
        float w = dinv[s] * dinv[d];
        const float* hs = h   + (size_t)s * 128 + lane * 4;
        float*       od = out + (size_t)d * 128 + lane * 4;
        float4 v = *(const float4*)hs;
        unsafeAtomicAdd(od + 0, w * v.x);
        unsafeAtomicAdd(od + 1, w * v.y);
        unsafeAtomicAdd(od + 2, w * v.z);
        unsafeAtomicAdd(od + 3, w * v.w);
    }
}

// ---------------------------------------------------------------------------
// global_max_pool (single graph): two-stage deterministic column max, O = 64
// ---------------------------------------------------------------------------
__global__ void gcn_pool_partial(const float* __restrict__ C,
                                 float* __restrict__ partial, int N, int O) {
    int col = threadIdx.x;          // blockDim.x == O
    float m = -INFINITY;
    for (int r = blockIdx.x; r < N; r += gridDim.x)
        m = fmaxf(m, C[(size_t)r * O + col]);
    partial[blockIdx.x * O + col] = m;
}

__global__ void gcn_pool_final(const float* __restrict__ partial,
                               float* __restrict__ out, int nb, int O) {
    int col = threadIdx.x;
    float m = -INFINITY;
    for (int b = 0; b < nb; ++b)
        m = fmaxf(m, partial[b * O + col]);
    out[col] = m;
}

// ---------------------------------------------------------------------------
// Host launcher
// ---------------------------------------------------------------------------
extern "C" void kernel_launch(void* const* d_in, const int* in_sizes, int n_in,
                              void* d_out, int out_size, void* d_ws, size_t ws_size,
                              hipStream_t stream) {
    const float* x    = (const float*)d_in[0];
    const int*   ei   = (const int*)  d_in[1];
    // d_in[2] = batch (all zeros, single graph) -> unused
    const float* W0   = (const float*)d_in[3];
    const float* b0   = (const float*)d_in[4];
    const float* W1   = (const float*)d_in[5];
    const float* b1   = (const float*)d_in[6];
    const float* W2   = (const float*)d_in[7];
    const float* b2   = (const float*)d_in[8];
    const float* Wlin = (const float*)d_in[9];
    const float* blin = (const float*)d_in[10];

    const int H = in_sizes[4];          // 128
    const int O = in_sizes[10];         // 64
    const int F = in_sizes[3] / H;      // 256
    const int N = in_sizes[0] / F;      // 50000
    const int E = in_sizes[1] / 2;      // 800000
    const int* src = ei;
    const int* dst = ei + E;

    // ---- workspace carve-up ----
    auto a256 = [](size_t v) { return (v + 255) & ~(size_t)255; };
    char* base = (char*)d_ws;
    size_t off = 0;
    float* dinv    = (float*)(base + off); off += a256((size_t)N * 4);
    float* bufA    = (float*)(base + off); off += a256((size_t)N * H * 4);
    float* bufB    = (float*)(base + off); off += a256((size_t)N * H * 4);
    float* bufC    = (float*)(base + off); off += a256((size_t)N * O * 4);
    __bf16* w0T    = (__bf16*)(base + off); off += a256((size_t)F * H * 2);
    __bf16* w1T    = (__bf16*)(base + off); off += a256((size_t)H * H * 2);
    __bf16* w2T    = (__bf16*)(base + off); off += a256((size_t)H * H * 2);
    __bf16* wlT    = (__bf16*)(base + off); off += a256((size_t)H * O * 2);
    const int NPB = 256;  // pool partial blocks
    float* partial = (float*)(base + off); off += a256((size_t)NPB * O * 4);
    (void)ws_size; (void)n_in; (void)out_size;

    // ---- 1. GCN normalization ----
    gcn_deg_init<<<(N + 255) / 256, 256, 0, stream>>>(dinv, N);
    gcn_deg_count<<<(E + 255) / 256, 256, 0, stream>>>(dst, dinv, E);
    gcn_rsqrt_inplace<<<(N + 255) / 256, 256, 0, stream>>>(dinv, N);

    // ---- 2. weight convert + transpose to bf16 ----
    gcn_w_convT<<<(F * H + 255) / 256, 256, 0, stream>>>(W0, w0T, F, H);
    gcn_w_convT<<<(H * H + 255) / 256, 256, 0, stream>>>(W1, w1T, H, H);
    gcn_w_convT<<<(H * H + 255) / 256, 256, 0, stream>>>(W2, w2T, H, H);
    gcn_w_convT<<<(H * O + 255) / 256, 256, 0, stream>>>(Wlin, wlT, H, O);

    const int gemmBlocks = (N + 127) / 128;
    const int aggBlocksI = (N * H + 255) / 256;
    const int aggBlocksE = (E + 7) / 8;

    // ---- layer 0: h = agg(x @ W0) + b0 ----
    gcn_gemm_bf16<128><<<gemmBlocks, 256, 0, stream>>>(x, w0T, nullptr, bufA, N, F);
    gcn_agg_init<<<aggBlocksI, 256, 0, stream>>>(bufA, dinv, b0, bufB, N);
    gcn_agg_edges<<<aggBlocksE, 256, 0, stream>>>(bufA, src, dst, dinv, bufB, E);

    // ---- layer 1 ----
    gcn_gemm_bf16<128><<<gemmBlocks, 256, 0, stream>>>(bufB, w1T, nullptr, bufA, N, H);
    gcn_agg_init<<<aggBlocksI, 256, 0, stream>>>(bufA, dinv, b1, bufB, N);
    gcn_agg_edges<<<aggBlocksE, 256, 0, stream>>>(bufA, src, dst, dinv, bufB, E);

    // ---- layer 2 ----
    gcn_gemm_bf16<128><<<gemmBlocks, 256, 0, stream>>>(bufB, w2T, nullptr, bufA, N, H);
    gcn_agg_init<<<aggBlocksI, 256, 0, stream>>>(bufA, dinv, b2, bufB, N);
    gcn_agg_edges<<<aggBlocksE, 256, 0, stream>>>(bufA, src, dst, dinv, bufB, E);

    // ---- final linear (+bias fused) ----
    gcn_gemm_bf16<64><<<gemmBlocks, 256, 0, stream>>>(bufB, wlT, blin, bufC, N, H);

    // ---- global max pool ----
    gcn_pool_partial<<<NPB, O, 0, stream>>>(bufC, partial, N, O);
    gcn_pool_final<<<1, O, 0, stream>>>(partial, (float*)d_out, NPB, O);
}
